// FusionHeadCoAtten_1288490189258
// MI455X (gfx1250) — matrix operand order repaired and checked by reading, not compiled, not run on hardware
//
#include <hip/hip_runtime.h>
#include <hip/hip_bf16.h>

// MI455X / gfx1250, wave32. D = B = 512.
// out row layout (2048 f32): [img | audio | C_img | C_audio]

typedef __attribute__((ext_vector_type(16))) _Float16 v16h;
typedef __attribute__((ext_vector_type(8)))  _Float16 v8h;
typedef __attribute__((ext_vector_type(8)))  float    v8f;

#define DD 512

// ---------------------------------------------------------------------------
// 1) convert img / W to f16 for WMMA (f32 accumulate keeps precision)
// ---------------------------------------------------------------------------
__global__ __launch_bounds__(256) void cvt_f16_kernel(const float* __restrict__ img,
                                                      const float* __restrict__ W,
                                                      _Float16* __restrict__ imgh,
                                                      _Float16* __restrict__ Wh) {
  int i = blockIdx.x * 256 + threadIdx.x;   // grid covers exactly 512*512
  imgh[i] = (_Float16)img[i];
  Wh[i]   = (_Float16)W[i];
}

// ---------------------------------------------------------------------------
// 2) u = tanh(img @ W^T + bias)  via v_wmma_f32_16x16x32_f16
//    One wave per 16x16 output tile; 8 waves (8 tiles) per block; 128 blocks.
//    A[m,k] = imgh[row m], B[k,n] = Wh[row n] (W^T access = row-major W reads)
// ---------------------------------------------------------------------------
__global__ __launch_bounds__(256) void gemm_tanh_kernel(const _Float16* __restrict__ Ah,
                                                        const _Float16* __restrict__ Bh,
                                                        const float* __restrict__ bias,
                                                        float* __restrict__ img_t) {
  const int wave = threadIdx.x >> 5;
  const int lane = threadIdx.x & 31;
  const int tile = blockIdx.x * 8 + wave;          // 0..1023
  const int tm   = (tile >> 5) << 4;               // batch-row base
  const int tn   = (tile & 31) << 4;               // feature-col base
  const int kh   = lane >> 4;                      // 0: lanes 0-15, 1: lanes 16-31
  const int l15  = lane & 15;

  const _Float16* arow = Ah + (size_t)(tm + l15) * DD;   // A rows: M = l15
  const _Float16* brow = Bh + (size_t)(tn + l15) * DD;   // B cols: N = l15

  v8f c = {};
  #pragma unroll
  for (int k0 = 0; k0 < DD; k0 += 32) {
    // A 16x32 f16 frag: lane holds K = kh*16 + [0..15] for row M = l15 (contiguous, matches B)
    v8h lo = *(const v8h*)(arow + k0 + kh * 16);
    v8h hi = *(const v8h*)(arow + k0 + kh * 16 + 8);
    v16h a;
    #pragma unroll
    for (int t = 0; t < 8; ++t) { a[t] = lo[t]; a[t + 8] = hi[t]; }
    // B 32x16 f16 frag: lane holds K = kh*16 + [0..15] for column N = l15 (contiguous)
    v16h bf = *(const v16h*)(brow + k0 + kh * 16);
    c = __builtin_amdgcn_wmma_f32_16x16x32_f16(false, a, false, bf,
                                               (short)0, c, false, false);
  }

  const float bn = bias[tn + l15];
  #pragma unroll
  for (int v = 0; v < 8; ++v) {
    int m = tm + kh * 8 + v;                       // C layout: VGPR v -> M = v (+8)
    img_t[(size_t)m * DD + tn + l15] = tanhf(c[v] + bn);
  }
}

// ---------------------------------------------------------------------------
// 3) rank-1 co-attention, one workgroup per batch.
//    2 exp passes over the 512x512 E matrix; 32x32 register tile per thread.
//    Dynamic LDS: u,a,wu,wa (4x512) + row partials (512x16) + col partials (512x16)
// ---------------------------------------------------------------------------
__global__ __launch_bounds__(256) void coatt_kernel(const float* __restrict__ img,
                                                    const float* __restrict__ audio,
                                                    const float* __restrict__ img_t,
                                                    float* __restrict__ out) {
  extern __shared__ float smem[];
  float* s_u    = smem;            // 512
  float* s_a    = smem + 512;      // 512
  float* s_wu   = smem + 1024;     // 512  u_i / rowsum_i
  float* s_wa   = smem + 1536;     // 512  a_j / colsum_j
  float* rowscr = smem + 2048;     // 512*16 row partials
  float* colscr = smem + 10240;    // 512*16 col partials

  const int b  = blockIdx.x;
  const int t  = threadIdx.x;
  const int tx = t & 15;           // owns cols j = tx + 16*jj
  const int ty = t >> 4;           // owns rows i = ty + 16*ii
  float* orow = out + (size_t)b * 2048;

  // concat copies + stage u, a in LDS
  for (int d = t; d < DD; d += 256) {
    float iv = img[(size_t)b * DD + d];
    float av = audio[(size_t)b * DD + d];
    orow[d]        = iv;
    orow[512 + d]  = av;
    s_a[d] = av;
    s_u[d] = img_t[(size_t)b * DD + d];
  }
  __syncthreads();

  float u_reg[32];
  #pragma unroll
  for (int ii = 0; ii < 32; ++ii) u_reg[ii] = s_u[ty + 16 * ii];

  // ---- pass 1: rowsum_i and colsum_j of E = exp(u_i * a_j) ----
  float rowp[32];
  #pragma unroll
  for (int ii = 0; ii < 32; ++ii) rowp[ii] = 0.f;

  for (int jj = 0; jj < 32; ++jj) {
    float aj = s_a[tx + 16 * jj];
    float cp = 0.f;
    #pragma unroll
    for (int ii = 0; ii < 32; ++ii) {
      float e = __expf(u_reg[ii] * aj);
      rowp[ii] += e;
      cp += e;
    }
    colscr[(tx + 16 * jj) * 16 + ty] = cp;      // stream col partial to LDS
  }
  #pragma unroll
  for (int ii = 0; ii < 32; ++ii) rowscr[(ty + 16 * ii) * 16 + tx] = rowp[ii];
  __syncthreads();

  // reduce 16 partials per row / per col
  for (int i = t; i < DD; i += 256) {
    float s = 0.f;
    #pragma unroll
    for (int x = 0; x < 16; ++x) s += rowscr[i * 16 + x];
    s_wu[i] = s_u[i] / s;
  }
  for (int j = t; j < DD; j += 256) {
    float s = 0.f;
    #pragma unroll
    for (int x = 0; x < 16; ++x) s += colscr[j * 16 + x];
    s_wa[j] = s_a[j] / s;
  }
  __syncthreads();

  // ---- pass 2: C_img_i = sum_j wa_j*E_ij ; C_audio_j = sum_i wu_i*E_ij ----
  float wu_reg[32];
  #pragma unroll
  for (int ii = 0; ii < 32; ++ii) wu_reg[ii] = s_wu[ty + 16 * ii];
  float cimg[32];
  #pragma unroll
  for (int ii = 0; ii < 32; ++ii) cimg[ii] = 0.f;

  for (int jj = 0; jj < 32; ++jj) {
    float aj  = s_a[tx + 16 * jj];
    float waj = s_wa[tx + 16 * jj];
    float ca  = 0.f;
    #pragma unroll
    for (int ii = 0; ii < 32; ++ii) {
      float e = __expf(u_reg[ii] * aj);
      cimg[ii] += waj * e;
      ca += wu_reg[ii] * e;
    }
    colscr[(tx + 16 * jj) * 16 + ty] = ca;      // overwrites pass-1 data (already consumed)
  }
  #pragma unroll
  for (int ii = 0; ii < 32; ++ii) rowscr[(ty + 16 * ii) * 16 + tx] = cimg[ii];
  __syncthreads();

  for (int i = t; i < DD; i += 256) {
    float s = 0.f;
    #pragma unroll
    for (int x = 0; x < 16; ++x) s += rowscr[i * 16 + x];
    orow[1024 + i] = s;
  }
  for (int j = t; j < DD; j += 256) {
    float s = 0.f;
    #pragma unroll
    for (int x = 0; x < 16; ++x) s += colscr[j * 16 + x];
    orow[1536 + j] = s;
  }
}

// ---------------------------------------------------------------------------
extern "C" void kernel_launch(void* const* d_in, const int* in_sizes, int n_in,
                              void* d_out, int out_size, void* d_ws, size_t ws_size,
                              hipStream_t stream) {
  const float* img   = (const float*)d_in[0];   // [512,512]
  const float* audio = (const float*)d_in[1];   // [512,512]
  const float* W     = (const float*)d_in[2];   // [512,512]
  const float* bias  = (const float*)d_in[3];   // [512]
  float* out = (float*)d_out;                   // [512,2048]

  char* ws = (char*)d_ws;
  _Float16* imgh  = (_Float16*)(ws);                        // 512 KB
  _Float16* Wh    = (_Float16*)(ws + (size_t)DD * DD * 2);  // 512 KB
  float*    img_t = (float*)   (ws + (size_t)DD * DD * 4);  // 1 MB

  cvt_f16_kernel <<<1024, 256, 0, stream>>>(img, W, imgh, Wh);
  gemm_tanh_kernel<<<128, 256, 0, stream>>>(imgh, Wh, bias, img_t);
  coatt_kernel   <<<512, 256, 18432 * sizeof(float), stream>>>(img, audio, img_t, out);
}
